// MPNN_block_seperate_6889127543478
// MI455X (gfx1250) — matrix operand order repaired and checked by reading, not compiled, not run on hardware
//
#include <hip/hip_runtime.h>
#include <math.h>

// ---------------------------------------------------------------------------
// MI455X / gfx1250 implementation (wave32, WMMA f16 -> f32 accumulate).
//   k_conv1   : implicit-GEMM conv1 (K=192) -> y1 fp32 [4096,128,32]
//   k_stats_cf: BN1 batch stats -> fused scale/shift
//   k_conv2   : implicit-GEMM conv2 (K=384), BN1+relu fused -> y2
//   k_stats_cf: BN2 batch stats
//   k_spa     : BN2+relu + faithful raw reshape -> spa f16 [512,256,128]
//   k_mapped  : spa @ map_w^T + b (WMMA) -> mapped f16
//   k_graph   : fused attention block: S=mapped@mapped^T, diag-mask, leaky,
//               softmax, +I, agg=P@spa, out=agg@theta^T+b.  adj never in HBM.
//   k_stats_cl, k_final : output BN + leaky (in place on d_out)
// WMMA fragments fed by ds_load_b128 pairs (padded LDS strides); pure-f16
// tile staging uses the CDNA5 async global->LDS pipe (ASYNCcnt).
// ---------------------------------------------------------------------------

typedef _Float16 half_t;
typedef __attribute__((ext_vector_type(16))) _Float16 v16h;
typedef __attribute__((ext_vector_type(8)))  _Float16 v8h;
typedef __attribute__((ext_vector_type(8)))  float    v8f;
typedef __attribute__((ext_vector_type(4)))  float    v4f;

#define WMMA_F16(acc, a, b)                                                    \
  acc = __builtin_amdgcn_wmma_f32_16x16x32_f16(false, (a), false, (b),         \
                                               (short)0, (acc), false, false)

// ---- CDNA5 async global->LDS copy (16B per lane, tracked by ASYNCcnt) ----
#define USE_ASYNC_LDS 1
#if USE_ASYNC_LDS
__device__ inline void async_copy_b128(half_t* lds_dst, const half_t* gsrc) {
  // LDS address operand = low 32 bits of the generic pointer (ISA 10.2:
  // flat->LDS uses addr[31:0]).
  const unsigned off = (unsigned)(unsigned long long)lds_dst;
  asm volatile("global_load_async_to_lds_b128 %0, %1, off"
               :: "v"(off), "v"(gsrc) : "memory");
}
__device__ inline void async_wait0() {
  asm volatile("s_wait_asynccnt 0x0" ::: "memory");
}
#endif

__device__ inline v8f zero8() {
  v8f z;
#pragma unroll
  for (int j = 0; j < 8; ++j) z[j] = 0.0f;
  return z;
}

// A fragment: 16x32 f16 tile, rows rbase..+15, K cols kb..+31, row-major src.
// lane(l&15)=row, K runs {h*16..h*16+15} (matches B fragment order).
__device__ inline v16h load_a_frag(const half_t* src, int ld, int rbase, int kb) {
  const int l = threadIdx.x & 31;
  const half_t* p = src + (size_t)(rbase + (l & 15)) * ld + kb + (l >> 4) * 16;
  const v8h lo = *(const v8h*)p;
  const v8h hi = *(const v8h*)(p + 8);
  return __builtin_shufflevector(lo, hi, 0, 1, 2, 3, 4, 5, 6, 7,
                                 8, 9, 10, 11, 12, 13, 14, 15);
}

// B fragment from K-transposed storage: logical B[k][n] = M[n][k], M row-major
// [N][K] (ld). Lane reads 16 contiguous halves -> two aligned 16B LDS loads.
__device__ inline v16h load_bT_frag(const half_t* src, int ld, int kb, int cbase) {
  const int l = threadIdx.x & 31;
  const half_t* p = src + (size_t)(cbase + (l & 15)) * ld + kb + (l >> 4) * 16;
  const v8h lo = *(const v8h*)p;
  const v8h hi = *(const v8h*)(p + 8);
  return __builtin_shufflevector(lo, hi, 0, 1, 2, 3, 4, 5, 6, 7,
                                 8, 9, 10, 11, 12, 13, 14, 15);
}

// Same but converting from fp32 global source (small weights, L2-resident).
__device__ inline v16h load_bT_frag_f32(const float* src, int ld, int kb, int cbase) {
  const int l = threadIdx.x & 31;
  const float* p = src + (size_t)(cbase + (l & 15)) * ld + kb + (l >> 4) * 16;
  v16h b;
#pragma unroll
  for (int q = 0; q < 4; ++q) {
    const v4f f = *(const v4f*)(p + 4 * q);
#pragma unroll
    for (int j = 0; j < 4; ++j) b[4 * q + j] = (half_t)f[j];
  }
  return b;
}

// ---------------------------------------------------------------------------
// conv1: Y[128,32] = W[128,192] * im2col(x)[192,32] per bn.
// Full K panel staged once.  Stride 200 halves (100 dw == 36 mod 64).
// ---------------------------------------------------------------------------
#define LDA1 200
__global__ __launch_bounds__(512, 1)
void k_conv1(const float* __restrict__ x, const float* __restrict__ w,
             float* __restrict__ y1) {
  __shared__ alignas(16) half_t As[128 * LDA1];  // [Cout][K] padded
  __shared__ alignas(16) half_t Bs[32 * LDA1];   // [T][K]   padded (im2col^T)
  const int bn = blockIdx.x;
  const int b = bn >> 8, n = bn & 255;
  const int tid = threadIdx.x;
  for (int i = tid; i < 128 * 192; i += 512) {
    const int m = i / 192, k = i - m * 192;
    const int dk = k >> 6, cin = k & 63;
    As[m * LDA1 + k] = (half_t)w[((size_t)m * 64 + cin) * 3 + dk];
  }
  for (int i = tid; i < 32 * 192; i += 512) {
    const int t = i / 192, k = i - t * 192;
    const int dk = k >> 6, cin = k & 63;
    const int tp = t + dk - 1;
    float v = 0.0f;
    if (tp >= 0 && tp < 32)
      v = x[(((size_t)b * 32 + tp) * 256 + n) * 64 + cin];
    Bs[t * LDA1 + k] = (half_t)v;
  }
  __syncthreads();
  const int wv = tid >> 5, mtile = wv >> 1, ntile = wv & 1;
  v8f acc = zero8();
#pragma unroll
  for (int kb = 0; kb < 192; kb += 32)
    WMMA_F16(acc, load_a_frag(As, LDA1, mtile * 16, kb),
                  load_bT_frag(Bs, LDA1, kb, ntile * 16));
  const int l = tid & 31, h = l >> 4, col = l & 15;
#pragma unroll
  for (int j = 0; j < 8; ++j)
    y1[((size_t)bn * 128 + mtile * 16 + j + 8 * h) * 32 + ntile * 16 + col] = acc[j];
}

// ---------------------------------------------------------------------------
// conv2: Cin=128, K=384 (stride 392 halves = 196 dw == 4 mod 64),
// BN1+relu fused on activation load.
// ---------------------------------------------------------------------------
#define LDA2 392
__global__ __launch_bounds__(512, 1)
void k_conv2(const float* __restrict__ y1, const float* __restrict__ w,
             const float* __restrict__ sc1, const float* __restrict__ sh1,
             float* __restrict__ y2) {
  __shared__ alignas(16) half_t As[128 * LDA2];
  __shared__ alignas(16) half_t Bs[32 * LDA2];
  const int bn = blockIdx.x;
  const int tid = threadIdx.x;
  for (int i = tid; i < 128 * 384; i += 512) {
    const int m = i / 384, k = i - m * 384;
    const int dk = k >> 7, cin = k & 127;
    As[m * LDA2 + k] = (half_t)w[((size_t)m * 128 + cin) * 3 + dk];
  }
  for (int i = tid; i < 32 * 384; i += 512) {
    const int t = i / 384, k = i - t * 384;
    const int dk = k >> 7, cin = k & 127;
    const int tp = t + dk - 1;
    float v = 0.0f;
    if (tp >= 0 && tp < 32) {
      v = sc1[cin] * y1[(size_t)bn * 4096 + cin * 32 + tp] + sh1[cin];
      v = v > 0.0f ? v : 0.0f;
    }
    Bs[t * LDA2 + k] = (half_t)v;
  }
  __syncthreads();
  const int wv = tid >> 5, mtile = wv >> 1, ntile = wv & 1;
  v8f acc = zero8();
#pragma unroll
  for (int kb = 0; kb < 384; kb += 32)
    WMMA_F16(acc, load_a_frag(As, LDA2, mtile * 16, kb),
                  load_bT_frag(Bs, LDA2, kb, ntile * 16));
  const int l = tid & 31, h = l >> 4, col = l & 15;
#pragma unroll
  for (int j = 0; j < 8; ++j)
    y2[((size_t)bn * 128 + mtile * 16 + j + 8 * h) * 32 + ntile * 16 + col] = acc[j];
}

// ---------------------------------------------------------------------------
// BN stats, layout [4096][128][32] (channel-first): block c owns channel c.
// ---------------------------------------------------------------------------
__global__ __launch_bounds__(256)
void k_stats_cf(const float* __restrict__ y, const float* __restrict__ g,
                const float* __restrict__ bt, float* __restrict__ sc,
                float* __restrict__ sh) {
  const int c = blockIdx.x;
  __shared__ float ssum[256], ssq[256];
  float s = 0.0f, q = 0.0f;
  for (int i = threadIdx.x; i < 4096 * 32; i += 256) {
    const int bn = i >> 5, t = i & 31;
    const float v = y[((size_t)bn * 128 + c) * 32 + t];
    s += v; q += v * v;
  }
  ssum[threadIdx.x] = s; ssq[threadIdx.x] = q;
  __syncthreads();
  for (int o = 128; o > 0; o >>= 1) {
    if (threadIdx.x < o) {
      ssum[threadIdx.x] += ssum[threadIdx.x + o];
      ssq[threadIdx.x]  += ssq[threadIdx.x + o];
    }
    __syncthreads();
  }
  if (threadIdx.x == 0) {
    const float cnt = 4096.0f * 32.0f;
    const float mean = ssum[0] / cnt;
    const float var = ssq[0] / cnt - mean * mean;
    const float scale = g[c] * rsqrtf(var + 1e-5f);
    sc[c] = scale;
    sh[c] = bt[c] - mean * scale;
  }
}

// BN stats, layout [131072][128] (channel-last) for the output BN.
__global__ __launch_bounds__(256)
void k_stats_cl(const float* __restrict__ y, const float* __restrict__ g,
                const float* __restrict__ bt, float* __restrict__ sc,
                float* __restrict__ sh) {
  const int c = blockIdx.x;
  __shared__ float ssum[256], ssq[256];
  float s = 0.0f, q = 0.0f;
  for (int i = threadIdx.x; i < 131072; i += 256) {
    const float v = y[(size_t)i * 128 + c];
    s += v; q += v * v;
  }
  ssum[threadIdx.x] = s; ssq[threadIdx.x] = q;
  __syncthreads();
  for (int o = 128; o > 0; o >>= 1) {
    if (threadIdx.x < o) {
      ssum[threadIdx.x] += ssum[threadIdx.x + o];
      ssq[threadIdx.x]  += ssq[threadIdx.x + o];
    }
    __syncthreads();
  }
  if (threadIdx.x == 0) {
    const float cnt = 131072.0f;
    const float mean = ssum[0] / cnt;
    const float var = ssq[0] / cnt - mean * mean;
    const float scale = g[c] * rsqrtf(var + 1e-5f);
    sc[c] = scale;
    sh[c] = bt[c] - mean * scale;
  }
}

// ---------------------------------------------------------------------------
// BN2 + relu + faithful raw reshape -> spa f16.  The reshape is the identity
// on flat offsets: spa[bp][n][c] = act(y2flat[bn*4096 + t*128 + c]).
// ---------------------------------------------------------------------------
__global__ __launch_bounds__(256)
void k_spa(const float* __restrict__ y2, const float* __restrict__ sc2,
           const float* __restrict__ sh2, half_t* __restrict__ spa) {
  const size_t v8i = (size_t)blockIdx.x * 256 + threadIdx.x;  // < 2^21
  const size_t i0 = v8i * 8;
  const int c0 = (int)(i0 & 127);
  const int n  = (int)((i0 >> 7) & 255);
  const int bp = (int)(i0 >> 15);
  const int b = bp >> 5, t = bp & 31;
  const int flat = t * 128 + c0;       // 8-aligned -> same ch for all 8
  const int ch = flat >> 5;
  const size_t bn = (size_t)b * 256 + n;
  const float scale = sc2[ch], shift = sh2[ch];
  const float* src = y2 + bn * 4096 + flat;
  const v4f f0 = *(const v4f*)src;
  const v4f f1 = *(const v4f*)(src + 4);
  v8h o;
#pragma unroll
  for (int j = 0; j < 4; ++j) {
    float v = scale * f0[j] + shift; v = v > 0.0f ? v : 0.0f; o[j] = (half_t)v;
  }
#pragma unroll
  for (int j = 0; j < 4; ++j) {
    float v = scale * f1[j] + shift; v = v > 0.0f ? v : 0.0f; o[4 + j] = (half_t)v;
  }
  *(v8h*)(spa + i0) = o;
}

// ---------------------------------------------------------------------------
// mapped = spa @ map_w^T + map_b -> f16.  Block: 128 rows.  Padded ld 136.
// spa tile staged via async global->LDS; weights converted in-VALU.
// ---------------------------------------------------------------------------
#define LDM 136
__global__ __launch_bounds__(512, 1)
void k_mapped(const half_t* __restrict__ spa, const float* __restrict__ mw,
              const float* __restrict__ mb, half_t* __restrict__ mapped) {
  __shared__ alignas(16) half_t Xs[128 * LDM];
  __shared__ alignas(16) half_t Ws[128 * LDM];
  const size_t row0 = (size_t)blockIdx.x * 128;
  const int tid = threadIdx.x;
  for (int i = tid; i < 2048; i += 512) {
    const int n = i >> 4, c0 = (i & 15) * 8;
#if USE_ASYNC_LDS
    async_copy_b128(Xs + n * LDM + c0, spa + (row0 + n) * 128 + c0);
#else
    *(v8h*)(Xs + n * LDM + c0) = *(const v8h*)(spa + (row0 + n) * 128 + c0);
#endif
    const float* wp = mw + (size_t)n * 128 + c0;
    const v4f f0 = *(const v4f*)wp;
    const v4f f1 = *(const v4f*)(wp + 4);
    v8h hw;
#pragma unroll
    for (int j = 0; j < 4; ++j) { hw[j] = (half_t)f0[j]; hw[4 + j] = (half_t)f1[j]; }
    *(v8h*)(Ws + n * LDM + c0) = hw;
  }
#if USE_ASYNC_LDS
  async_wait0();
#endif
  __syncthreads();
  const int wv = tid >> 5;
  const int msub = (wv & 7) * 16;
  const int nbase = (wv >> 3) * 64;
  v8f acc[4];
#pragma unroll
  for (int nt = 0; nt < 4; ++nt) acc[nt] = zero8();
#pragma unroll
  for (int kb = 0; kb < 128; kb += 32) {
    const v16h a = load_a_frag(Xs, LDM, msub, kb);
#pragma unroll
    for (int nt = 0; nt < 4; ++nt)
      WMMA_F16(acc[nt], a, load_bT_frag(Ws, LDM, kb, nbase + nt * 16));
  }
  const int l = tid & 31, h = l >> 4, col = l & 15;
#pragma unroll
  for (int nt = 0; nt < 4; ++nt)
#pragma unroll
    for (int j = 0; j < 8; ++j) {
      const int r = msub + j + 8 * h;
      const int cc = nbase + nt * 16 + col;
      mapped[(row0 + r) * 128 + cc] = (half_t)(acc[nt][j] + mb[cc]);
    }
}

// ---------------------------------------------------------------------------
// Fused per-graph block (one workgroup / bp).  16 waves x one 16-row tile.
//   Ms : mapped [256][136] padded  (A and B^T for scores) -- async staged
//   SsT: spa^T  [128][264] padded  (B^T for agg; k = node index)
//   Ps : per-wave C->A relayout scratch, row stride 40 (conflict-free b128)
// 154KB LDS per workgroup (CDNA5 320KB WGP LDS).
// ---------------------------------------------------------------------------
#define LDT 264
#define LDP 40
__global__ __launch_bounds__(512, 1)
void k_graph(const half_t* __restrict__ spa, const half_t* __restrict__ mapped,
             const float* __restrict__ tw, const float* __restrict__ tb,
             float* __restrict__ outraw) {
  __shared__ alignas(16) half_t Ms[256 * LDM];
  __shared__ alignas(16) half_t SsT[128 * LDT];
  __shared__ alignas(16) half_t Ps[16][16 * LDP];
  const int bp = blockIdx.x;
  const int tid = threadIdx.x;
  const size_t base = (size_t)bp * 256 * 128;
  for (int i = tid; i < 4096; i += 512) {
    const int n = i >> 4, c0 = (i & 15) * 8;
#if USE_ASYNC_LDS
    async_copy_b128(Ms + n * LDM + c0, mapped + base + (size_t)n * 128 + c0);
#else
    *(v8h*)(Ms + n * LDM + c0) = *(const v8h*)(mapped + base + (size_t)n * 128 + c0);
#endif
    const v8h vv = *(const v8h*)(spa + base + (size_t)n * 128 + c0);
#pragma unroll
    for (int j = 0; j < 8; ++j) SsT[(c0 + j) * LDT + n] = vv[j];
  }
#if USE_ASYNC_LDS
  async_wait0();
#endif
  __syncthreads();

  const int wv = tid >> 5;
  const int r0 = wv * 16;
  const int l = tid & 31, h = l >> 4, col = l & 15;

  // ---- scores: 16 rows x 256 cols in 16 f32 accum tiles ----
  v8f s[16];
#pragma unroll
  for (int nt = 0; nt < 16; ++nt) s[nt] = zero8();
#pragma unroll
  for (int kb = 0; kb < 128; kb += 32) {
    const v16h a = load_a_frag(Ms, LDM, r0, kb);
#pragma unroll
    for (int nt = 0; nt < 16; ++nt)
      WMMA_F16(s[nt], a, load_bT_frag(Ms, LDM, kb, nt * 16));
  }

  // ---- diag-mask, leaky, row softmax (shfl within 16-lane halves), +I ----
#pragma unroll
  for (int j = 0; j < 8; ++j) {
    const int grow = r0 + j + 8 * h;
    float mx = -3.0e38f;
#pragma unroll
    for (int nt = 0; nt < 16; ++nt) {
      const int gcol = nt * 16 + col;
      float v = s[nt][j];
      if (gcol == grow) v -= 1.0e8f;
      v = v > 0.0f ? v : 0.01f * v;
      s[nt][j] = v;
      mx = fmaxf(mx, v);
    }
#pragma unroll
    for (int d = 1; d < 16; d <<= 1) mx = fmaxf(mx, __shfl_xor(mx, d, 32));
    float sm = 0.0f;
#pragma unroll
    for (int nt = 0; nt < 16; ++nt) {
      const float e = __expf(s[nt][j] - mx);
      s[nt][j] = e;
      sm += e;
    }
#pragma unroll
    for (int d = 1; d < 16; d <<= 1) sm += __shfl_xor(sm, d, 32);
    const float inv = 1.0f / sm;
#pragma unroll
    for (int nt = 0; nt < 16; ++nt) {
      const int gcol = nt * 16 + col;
      float p = s[nt][j] * inv;
      if (gcol == grow) p += 1.0f;
      s[nt][j] = p;
    }
  }

  // ---- agg = P @ spa : K = N = 256 in 8 chunks of 32 ----
  half_t* myP = &Ps[wv][0];
  v8f agg[8];
#pragma unroll
  for (int ct = 0; ct < 8; ++ct) agg[ct] = zero8();
  for (int kk = 0; kk < 8; ++kk) {
#pragma unroll
    for (int hf = 0; hf < 2; ++hf) {
      const int nt = kk * 2 + hf;
#pragma unroll
      for (int j = 0; j < 8; ++j)
        myP[(j + 8 * h) * LDP + hf * 16 + col] = (half_t)s[nt][j];
    }
    __builtin_amdgcn_wave_barrier();   // order per-wave LDS write -> read
    const v16h a = load_a_frag(myP, LDP, 0, 0);
#pragma unroll
    for (int ct = 0; ct < 8; ++ct)
      WMMA_F16(agg[ct], a, load_bT_frag(SsT, LDT, kk * 32, ct * 16));
    __builtin_amdgcn_wave_barrier();
  }

  // ---- out = agg @ theta^T : K = 128 in 4 chunks of 32 ----
  v8f o[8];
#pragma unroll
  for (int ot = 0; ot < 8; ++ot) o[ot] = zero8();
  for (int kk = 0; kk < 4; ++kk) {
#pragma unroll
    for (int hf = 0; hf < 2; ++hf) {
      const int ct = kk * 2 + hf;
#pragma unroll
      for (int j = 0; j < 8; ++j)
        myP[(j + 8 * h) * LDP + hf * 16 + col] = (half_t)agg[ct][j];
    }
    __builtin_amdgcn_wave_barrier();
    const v16h a = load_a_frag(myP, LDP, 0, 0);
#pragma unroll
    for (int ot = 0; ot < 8; ++ot)
      WMMA_F16(o[ot], a, load_bT_frag_f32(tw, 128, kk * 32, ot * 16));
    __builtin_amdgcn_wave_barrier();
  }

#pragma unroll
  for (int ot = 0; ot < 8; ++ot)
#pragma unroll
    for (int j = 0; j < 8; ++j) {
      const int r = r0 + j + 8 * h;
      const int oc = ot * 16 + col;
      outraw[base + (size_t)r * 128 + oc] = o[ot][j] + tb[oc];
    }
}

// ---------------------------------------------------------------------------
// Final in-place BN + leaky_relu over d_out (channel-last), 4-wide.
// ---------------------------------------------------------------------------
__global__ __launch_bounds__(256)
void k_final(float* __restrict__ out, const float* __restrict__ sc,
             const float* __restrict__ sh) {
  const size_t i0 = ((size_t)blockIdx.x * 256 + threadIdx.x) * 4;
  const int c0 = (int)(i0 & 127);
  v4f v = *(const v4f*)(out + i0);
#pragma unroll
  for (int j = 0; j < 4; ++j) {
    const float t = sc[c0 + j] * v[j] + sh[c0 + j];
    v[j] = t > 0.0f ? t : 0.01f * t;
  }
  *(v4f*)(out + i0) = v;
}

// ---------------------------------------------------------------------------
extern "C" void kernel_launch(void* const* d_in, const int* in_sizes, int n_in,
                              void* d_out, int out_size, void* d_ws,
                              size_t ws_size, hipStream_t stream) {
  const float* x       = (const float*)d_in[0];
  const float* conv1_w = (const float*)d_in[1];
  const float* bn1_g   = (const float*)d_in[2];
  const float* bn1_b   = (const float*)d_in[3];
  const float* conv2_w = (const float*)d_in[4];
  const float* bn2_g   = (const float*)d_in[5];
  const float* bn2_b   = (const float*)d_in[6];
  const float* map_w   = (const float*)d_in[7];
  const float* map_b   = (const float*)d_in[8];
  const float* theta_w = (const float*)d_in[9];
  const float* theta_b = (const float*)d_in[10];
  const float* bnS_g   = (const float*)d_in[11];
  const float* bnS_b   = (const float*)d_in[12];
  float* out = (float*)d_out;

  char* ws = (char*)d_ws;
  float*  y1     = (float*)(ws + 0);                      // 64 MiB
  float*  y2     = (float*)(ws + (size_t)67108864);       // 64 MiB
  half_t* spa    = (half_t*)(ws + (size_t)134217728);     // 32 MiB
  half_t* mapped = (half_t*)(ws + (size_t)167772160);     // 32 MiB
  float*  sc1    = (float*)(ws + (size_t)201326592);
  float*  sh1 = sc1 + 128;
  float*  sc2 = sc1 + 256;
  float*  sh2 = sc1 + 384;
  float*  scS = sc1 + 512;
  float*  shS = sc1 + 640;

  k_conv1<<<4096, 512, 0, stream>>>(x, conv1_w, y1);
  k_stats_cf<<<128, 256, 0, stream>>>(y1, bn1_g, bn1_b, sc1, sh1);
  k_conv2<<<4096, 512, 0, stream>>>(y1, conv2_w, sc1, sh1, y2);
  k_stats_cf<<<128, 256, 0, stream>>>(y2, bn2_g, bn2_b, sc2, sh2);
  k_spa<<<8192, 256, 0, stream>>>(y2, sc2, sh2, spa);
  k_mapped<<<1024, 512, 0, stream>>>(spa, map_w, map_b, mapped);
  k_graph<<<512, 512, 0, stream>>>(spa, mapped, theta_w, theta_b, out);
  k_stats_cl<<<128, 256, 0, stream>>>(out, bnS_g, bnS_b, scS, shS);
  k_final<<<16384, 256, 0, stream>>>(out, scS, shS);
}
